// FusionLayer_85899345920421
// MI455X (gfx1250) — matrix operand order repaired and checked by reading, not compiled, not run on hardware
//
#include <hip/hip_runtime.h>
#include <hip/hip_bf16.h>

typedef __attribute__((ext_vector_type(16))) _Float16 v16h;
typedef __attribute__((ext_vector_type(8)))  _Float16 v8h;
typedef __attribute__((ext_vector_type(8)))  float    v8f;

#define M_PTS   200000
#define N_CAM   6
#define C_IMG   96
#define H_FEAT  32
#define W_FEAT  88
#define PTS_DIM 128
#define FUSE_C  224            // 128 + 96
#define GZ      32
#define GY      512
#define GX      448
#define GRID_N  (GZ*GY*GX)     // 7340032
#define NK      27
#define KSTEPS  7              // 224 / 32
#define NTILES  8              // 128 / 16
#define MTILES  (M_PTS/16)     // 12500
#define NWAVES  (MTILES)       // (MTILES/2 pairs) * 2 N-halves = 12500 waves
#define WAVES_PER_BLOCK 8
#define BLOCK_THREADS   (WAVES_PER_BLOCK*32)
#define BPK     (KSTEPS*NTILES*32*32)   // bytes of packed B per k offset = 57344
#define COPY_CHUNKS (BPK/16/BLOCK_THREADS) // 14 x b128 per thread per k

// ---------------- workspace layout (bytes) ----------------
#define WS_XFORM  ((size_t)0)                       // 6*24 floats
#define WS_GXGY   ((size_t)1024)                    // M*6*2 floats = 9.6 MB
#define WS_FUSE   ((size_t)9601024)                 // (M+16)*224 f16 = 89.6 MB
#define WS_GRID   ((size_t)99208192)                // 29.36 MB int32
#define WS_BPACK  ((size_t)128568320)               // 27*7*8*32*16 f16 = 1.55 MB

// ============================================================
// Kernel 0: fold lidar_aug^-1 into per-camera transforms
// ============================================================
__global__ void prep_xform(const float* __restrict__ l2i,
                           const float* __restrict__ iaug,
                           const float* __restrict__ laug,
                           float* __restrict__ xf) {
    if (threadIdx.x != 0 || blockIdx.x != 0) return;
    float L[9], t[3];
    for (int i = 0; i < 3; ++i) {
        for (int j = 0; j < 3; ++j) L[i*3+j] = laug[i*4+j];
        t[i] = laug[i*4+3];
    }
    float inv[9];
    float det = L[0]*(L[4]*L[8]-L[5]*L[7]) - L[1]*(L[3]*L[8]-L[5]*L[6])
              + L[2]*(L[3]*L[7]-L[4]*L[6]);
    float id = 1.0f / det;
    inv[0] =  (L[4]*L[8]-L[5]*L[7])*id;
    inv[1] = -(L[1]*L[8]-L[2]*L[7])*id;
    inv[2] =  (L[1]*L[5]-L[2]*L[4])*id;
    inv[3] = -(L[3]*L[8]-L[5]*L[6])*id;
    inv[4] =  (L[0]*L[8]-L[2]*L[6])*id;
    inv[5] = -(L[0]*L[5]-L[2]*L[3])*id;
    inv[6] =  (L[3]*L[7]-L[4]*L[6])*id;
    inv[7] = -(L[0]*L[7]-L[1]*L[6])*id;
    inv[8] =  (L[0]*L[4]-L[1]*L[3])*id;
    for (int n = 0; n < N_CAM; ++n) {
        const float* Rm = l2i + n*16;
        float* o = xf + n*24;
        float A1[9];
        for (int i = 0; i < 3; ++i)
            for (int j = 0; j < 3; ++j) {
                float s = 0.f;
                for (int q = 0; q < 3; ++q) s += Rm[i*4+q]*inv[q*3+j];
                A1[i*3+j] = s;
            }
        for (int i = 0; i < 9; ++i) o[i] = A1[i];
        for (int i = 0; i < 3; ++i) {
            float s = Rm[i*4+3];
            for (int j = 0; j < 3; ++j) s -= A1[i*3+j]*t[j];
            o[9+i] = s;
        }
        const float* Gm = iaug + n*16;
        for (int i = 0; i < 3; ++i)
            for (int j = 0; j < 3; ++j) o[12 + i*3+j] = Gm[i*4+j];
        for (int i = 0; i < 3; ++i) o[21+i] = Gm[i*4+3];
    }
}

// ============================================================
// Kernel 1: pre-pack conv_w (27,224,128) f32 -> f16 B-fragments
// index = ((k*7+kk)*8 + t)*32 + lane ; 16 halves each
// ============================================================
__global__ void prep_bpack(const float* __restrict__ w,
                           _Float16* __restrict__ bpack) {
    int idx = blockIdx.x * blockDim.x + threadIdx.x;     // 0..48383
    if (idx >= NK*KSTEPS*NTILES*32) return;
    int lane = idx & 31;
    int tmp  = idx >> 5;
    int t    = tmp & 7;
    int tmp2 = tmp >> 3;
    int kk   = tmp2 % KSTEPS;
    int k    = tmp2 / KSTEPS;
    int col   = t*16 + (lane & 15);
    int kbase = kk*32 + ((lane >> 4) * 16);
    _Float16* o = bpack + (size_t)idx * 16;
    #pragma unroll
    for (int j = 0; j < 16; ++j)
        o[j] = (_Float16)w[((size_t)(k*FUSE_C + kbase + j))*PTS_DIM + col];
}

// ============================================================
// Kernel 2: projection -> (gx, gy) per (point, cam)
// ============================================================
__global__ void proj_kernel(const int* __restrict__ vc,
                            const float* __restrict__ xf,
                            float* __restrict__ gxgy) {
    int idx = blockIdx.x * blockDim.x + threadIdx.x;
    if (idx >= M_PTS * N_CAM) return;
    int m = idx / N_CAM, cam = idx % N_CAM;
    float x = (float)vc[m*4+3];
    float y = (float)vc[m*4+2];
    float z = (float)vc[m*4+1];
    const float* f = xf + cam*24;
    float c0 = f[0]*x + f[1]*y + f[2]*z + f[9];
    float c1 = f[3]*x + f[4]*y + f[5]*z + f[10];
    float c2 = f[6]*x + f[7]*y + f[8]*z + f[11];
    float zz = fminf(fmaxf(c2, 1e-5f), 100000.0f);
    float vx = c0 / zz, vy = c1 / zz;
    float d0 = f[12]*vx + f[13]*vy + f[14]*zz + f[21];
    float d1 = f[15]*vx + f[16]*vy + f[17]*zz + f[22];
    gxgy[idx*2+0] = (d0 / 704.0f - 0.5f) * 2.0f;
    gxgy[idx*2+1] = (d1 / 256.0f - 0.5f) * 2.0f;
}

// ============================================================
// Kernel 3: fuse rows f16: [pts(128) | sampled(96)]
// ============================================================
__device__ __forceinline__ float corner_tap(const float* __restrict__ im,
                                            float xi, float yi, float w) {
    bool valid = (xi >= 0.f) & (xi <= (float)(W_FEAT-1)) &
                 (yi >= 0.f) & (yi <= (float)(H_FEAT-1));
    int xc = (int)fminf(fmaxf(xi, 0.f), (float)(W_FEAT-1));
    int yc = (int)fminf(fmaxf(yi, 0.f), (float)(H_FEAT-1));
    return valid ? im[yc*W_FEAT + xc] * w : 0.0f;
}

__global__ void fuse_kernel(const float* __restrict__ pts,
                            const float* __restrict__ img,
                            const float* __restrict__ gxgy,
                            _Float16* __restrict__ fuse) {
    int m = blockIdx.x;
    int tid = threadIdx.x;
    _Float16* frow = fuse + (size_t)m * FUSE_C;
    if (tid < PTS_DIM) {
        frow[tid] = (_Float16)pts[(size_t)m*PTS_DIM + tid];
    } else {
        int c = tid - PTS_DIM;
        float s = 0.f;
        #pragma unroll
        for (int cam = 0; cam < N_CAM; ++cam) {
            float gx = gxgy[(m*N_CAM + cam)*2 + 0];
            float gy = gxgy[(m*N_CAM + cam)*2 + 1];
            float x = (gx + 1.0f) * 0.5f * (float)(W_FEAT-1);
            float y = (gy + 1.0f) * 0.5f * (float)(H_FEAT-1);
            float x0 = floorf(x), y0 = floorf(y);
            float wx1 = x - x0, wx0 = 1.0f - wx1;
            float wy1 = y - y0, wy0 = 1.0f - wy1;
            const float* im = img + ((size_t)(cam*C_IMG + c)) * (H_FEAT*W_FEAT);
            s += corner_tap(im, x0,      y0,      wx0*wy0);
            s += corner_tap(im, x0+1.f,  y0,      wx1*wy0);
            s += corner_tap(im, x0,      y0+1.f,  wx0*wy1);
            s += corner_tap(im, x0+1.f,  y0+1.f,  wx1*wy1);
        }
        frow[PTS_DIM + c] = (_Float16)s;
    }
}

// ============================================================
// grid init (-1), scatter, zero row M
// ============================================================
__global__ void grid_init(int* __restrict__ grid) {
    int idx = blockIdx.x * blockDim.x + threadIdx.x;
    if (idx < GRID_N) grid[idx] = -1;
}
__global__ void grid_scatter(const int* __restrict__ vc, int* __restrict__ grid) {
    int m = blockIdx.x * blockDim.x + threadIdx.x;
    if (m >= M_PTS) return;
    int cz = vc[m*4+1], cy = vc[m*4+2], cx = vc[m*4+3];
    grid[(cz*GY + cy)*GX + cx] = m;
}
__global__ void zero_row(_Float16* __restrict__ fuse) {
    int tid = threadIdx.x;
    if (tid < FUSE_C) fuse[(size_t)M_PTS * FUSE_C + tid] = (_Float16)0.0f;
}

// ============================================================
// CDNA5 async copy: global -> LDS, ASYNCcnt-tracked (no VGPR data)
// ============================================================
__device__ __forceinline__ void async_copy16(unsigned lds_byte, unsigned goff,
                                             unsigned long long gbase) {
    asm volatile("global_load_async_to_lds_b128 %0, %1, %2"
                 :: "v"(lds_byte), "v"(goff), "s"(gbase)
                 : "memory");
}
__device__ __forceinline__ void wait_async0() {
    asm volatile("s_wait_asynccnt 0x0" ::: "memory");
}

__device__ __forceinline__ v16h loadA(const _Float16* __restrict__ aRow, int kk) {
    v8h lo = *(const v8h*)(aRow + kk*32);
    v8h hi = *(const v8h*)(aRow + kk*32 + 8);
    return __builtin_shufflevector(lo, hi,
            0,1,2,3,4,5,6,7,8,9,10,11,12,13,14,15);
}

// B fragment read with the XOR-16 bank swizzle (sw = 0 or 16 per lane)
__device__ __forceinline__ v16h loadB(const char* lanePtr, int frag, unsigned sw) {
    v8h lo = *(const v8h*)(lanePtr + frag*1024 + sw);
    v8h hi = *(const v8h*)(lanePtr + frag*1024 + (16u ^ sw));
    return __builtin_shufflevector(lo, hi,
            0,1,2,3,4,5,6,7,8,9,10,11,12,13,14,15);
}

// stencil-tap -> fuse row index (row M_PTS is the zero row)
__device__ __forceinline__ int stencil_row(int k, int cz, int cy, int cx,
                                           const int* __restrict__ grid) {
    const int dz = k / 9 - 1;
    const int dy = (k / 3) % 3 - 1;
    const int dx = k % 3 - 1;
    const int nz = cz + dz, ny = cy + dy, nx = cx + dx;
    const bool valid = (nz >= 0) & (nz < GZ) & (ny >= 0) & (ny < GY) &
                       (nx >= 0) & (nx < GX);
    const int zc = min(max(nz, 0), GZ-1);
    const int yc = min(max(ny, 0), GY-1);
    const int xc = min(max(nx, 0), GX-1);
    const int nid = grid[(zc*GY + yc)*GX + xc];
    return (valid && nid >= 0) ? nid : M_PTS;
}

// ============================================================
// Kernel 5: sparse-conv GEMM via WMMA f32 <- f16 x f16
// wave = 32 points (2 M-tiles) x 64 cols (4 N-tiles): each LDS B
// fragment feeds TWO WMMAs -> ds:wmma = 1:1 (was 2:1).
// 8 waves/block share double-buffered, bank-swizzled B in LDS
// filled by the CDNA5 async-to-LDS DMA pipeline.
// ============================================================
__launch_bounds__(BLOCK_THREADS)
__global__ void gemm_kernel(const int* __restrict__ vc,
                            const _Float16* __restrict__ fuse,
                            const _Float16* __restrict__ bpack,
                            const int* __restrict__ grid,
                            const float* __restrict__ bn_gamma,
                            const float* __restrict__ bn_beta,
                            const float* __restrict__ bn_mean,
                            const float* __restrict__ bn_var,
                            float* __restrict__ out) {
    extern __shared__ __align__(128) char smem[];   // 2 * BPK bytes
    const int tid  = threadIdx.x;
    const int lane = tid & 31;
    int gwave = blockIdx.x * WAVES_PER_BLOCK + (tid >> 5);
    gwave = gwave < NWAVES ? gwave : NWAVES - 1;    // dup-write clamp (idempotent)
    const int pair  = gwave >> 1;                   // 0..6249  (32-row block)
    const int nhalf = gwave & 1;                    // 0..1     (64-col half)
    const int rowBase = pair * 32;
    const int m0 = rowBase + (lane & 15);
    const int m1 = m0 + 16;

    const int cz0 = vc[m0*4+1], cy0 = vc[m0*4+2], cx0 = vc[m0*4+3];
    const int cz1 = vc[m1*4+1], cy1 = vc[m1*4+2], cx1 = vc[m1*4+3];
    const int koff = (lane >> 4) * 16;
    const unsigned sw = (lane & 8) ? 16u : 0u;      // bank swizzle bit

    const unsigned ldsBase = (unsigned)(uintptr_t)smem;
    const unsigned long long gB = (unsigned long long)(uintptr_t)bpack;

    // prefill buffer 0 with B[k=0] (write side applies the same swizzle)
    #pragma unroll
    for (int j = 0; j < COPY_CHUNKS; ++j) {
        unsigned g = (unsigned)(j*BLOCK_THREADS + tid) * 16u;
        unsigned d = g ^ ((g >> 4) & 16u);          // flip bit4 when bit8 set
        async_copy16(ldsBase + d, g, gB);
    }

    // prefetch tap 0 rows + first A fragments (overlaps the async prefill)
    int row0 = stencil_row(0, cz0, cy0, cx0, grid);
    int row1 = stencil_row(0, cz1, cy1, cx1, grid);
    const _Float16* aRow0 = fuse + (unsigned)(row0 * FUSE_C) + koff;
    const _Float16* aRow1 = fuse + (unsigned)(row1 * FUSE_C) + koff;
    v16h a0 = loadA(aRow0, 0);
    v16h a1 = loadA(aRow1, 0);

    wait_async0();
    __syncthreads();

    v8f acc0[4], acc1[4];
    #pragma unroll
    for (int t = 0; t < 4; ++t) {
        acc0[t] = (v8f){0.f,0.f,0.f,0.f,0.f,0.f,0.f,0.f};
        acc1[t] = (v8f){0.f,0.f,0.f,0.f,0.f,0.f,0.f,0.f};
    }

    for (int k = 0; k < NK; ++k) {
        // async copy of B[k+1] overlaps the whole compute phase of B[k]
        if (k < NK-1) {
            unsigned dstBase = ldsBase + (unsigned)(((k+1)&1) * BPK);
            unsigned long long gsrc = gB + (unsigned long long)(k+1) * BPK;
            #pragma unroll
            for (int j = 0; j < COPY_CHUNKS; ++j) {
                unsigned g = (unsigned)(j*BLOCK_THREADS + tid) * 16u;
                unsigned d = g ^ ((g >> 4) & 16u);
                async_copy16(dstBase + d, g, gsrc);
            }
        }

        // prefetch next tap's gather rows (grid loads overlap compute)
        const int row0n = (k < NK-1) ? stencil_row(k+1, cz0, cy0, cx0, grid) : row0;
        const int row1n = (k < NK-1) ? stencil_row(k+1, cz1, cy1, cx1, grid) : row1;
        const _Float16* aRow0N = fuse + (unsigned)(row0n * FUSE_C) + koff;
        const _Float16* aRow1N = fuse + (unsigned)(row1n * FUSE_C) + koff;

        const char* lanePtr = smem + (k&1)*BPK + lane*32;

        #pragma unroll
        for (int kk = 0; kk < KSTEPS; ++kk) {
            // batch this wave's 4 B fragments (its 64-col half) into regs
            v16h breg[4];
            #pragma unroll
            for (int t = 0; t < 4; ++t)
                breg[t] = loadB(lanePtr, kk*NTILES + nhalf*4 + t, sw);
            // prefetch next A fragments (next kk, or first of next tap)
            v16h a0n = (kk < KSTEPS-1) ? loadA(aRow0, kk+1) : loadA(aRow0N, 0);
            v16h a1n = (kk < KSTEPS-1) ? loadA(aRow1, kk+1) : loadA(aRow1N, 0);
            #pragma unroll
            for (int t = 0; t < 4; ++t) {
                acc0[t] = __builtin_amdgcn_wmma_f32_16x16x32_f16(
                            false, a0, false, breg[t], (short)0, acc0[t],
                            false, false);
                acc1[t] = __builtin_amdgcn_wmma_f32_16x16x32_f16(
                            false, a1, false, breg[t], (short)0, acc1[t],
                            false, false);
            }
            a0 = a0n;
            a1 = a1n;
        }
        row0 = row0n; aRow0 = aRow0N;
        row1 = row1n; aRow1 = aRow1N;

        if (k < NK-1) {
            wait_async0();       // this wave's B[k+1] chunks landed in LDS
            __syncthreads();     // all waves' chunks landed -> buffer ready
        }
    }

    // fused BN + ReLU epilogue; C layout: VGPR r -> row r + 8*(lane>=16), col lane&15
    const int colBase = nhalf*64 + (lane & 15);
    const int rb = (lane >> 4) * 8;
    #pragma unroll
    for (int t = 0; t < 4; ++t) {
        const int col = colBase + t*16;
        const float mean = bn_mean[col];
        const float rs   = rsqrtf(bn_var[col] + 1e-5f);
        const float ga   = bn_gamma[col];
        const float be   = bn_beta[col];
        #pragma unroll
        for (int r = 0; r < 8; ++r) {
            float v0 = (acc0[t][r] - mean) * rs * ga + be;
            out[(size_t)(rowBase + rb + r) * PTS_DIM + col] = fmaxf(v0, 0.0f);
            float v1 = (acc1[t][r] - mean) * rs * ga + be;
            out[(size_t)(rowBase + 16 + rb + r) * PTS_DIM + col] = fmaxf(v1, 0.0f);
        }
    }
}

// ============================================================
extern "C" void kernel_launch(void* const* d_in, const int* in_sizes, int n_in,
                              void* d_out, int out_size, void* d_ws, size_t ws_size,
                              hipStream_t stream) {
    const float* pts   = (const float*)d_in[0];
    const int*   vc    = (const int*)  d_in[1];
    const float* img   = (const float*)d_in[2];
    const float* l2i   = (const float*)d_in[3];
    const float* iaug  = (const float*)d_in[4];
    const float* laug  = (const float*)d_in[5];
    const float* w     = (const float*)d_in[6];
    const float* gamma = (const float*)d_in[7];
    const float* beta  = (const float*)d_in[8];
    const float* mean  = (const float*)d_in[9];
    const float* var   = (const float*)d_in[10];
    float* out = (float*)d_out;

    char* ws = (char*)d_ws;
    float*    xf    = (float*)   (ws + WS_XFORM);
    float*    gxgy  = (float*)   (ws + WS_GXGY);
    _Float16* fuse  = (_Float16*)(ws + WS_FUSE);
    int*      grid  = (int*)     (ws + WS_GRID);
    _Float16* bpack = (_Float16*)(ws + WS_BPACK);

    prep_xform<<<1, 1, 0, stream>>>(l2i, iaug, laug, xf);
    prep_bpack<<<(NK*KSTEPS*NTILES*32 + 127)/128, 128, 0, stream>>>(w, bpack);
    proj_kernel<<<(M_PTS*N_CAM + 255)/256, 256, 0, stream>>>(vc, xf, gxgy);
    grid_init<<<GRID_N/256, 256, 0, stream>>>(grid);
    grid_scatter<<<(M_PTS + 255)/256, 256, 0, stream>>>(vc, grid);
    zero_row<<<1, 256, 0, stream>>>(fuse);
    fuse_kernel<<<M_PTS, FUSE_C, 0, stream>>>(pts, img, gxgy, fuse);

    const int gemmBlocks = (NWAVES + WAVES_PER_BLOCK - 1) / WAVES_PER_BLOCK; // 1563
    gemm_kernel<<<gemmBlocks, BLOCK_THREADS, 2*BPK, stream>>>(
        vc, fuse, bpack, grid, gamma, beta, mean, var, out);
}
